// VectorQuantizer_21595095564368
// MI455X (gfx1250) — matrix operand order repaired and checked by reading, not compiled, not run on hardware
//
#include <hip/hip_runtime.h>
#include <stdint.h>

// ---------------------------------------------------------------------------
// VQ-VAE vector quantizer for MI455X (gfx1250, wave32, WMMA)
//   x:         [B=64, C=256, H=32, W=32] f32   (flat rows strided by 4KB!)
//   embedding: [K=4096, D=256] f32
//   out: quantized [B,C,H,W] f32, then loss, then perplexity
// ---------------------------------------------------------------------------

typedef __bf16 bf16;
typedef __attribute__((ext_vector_type(16))) __bf16 v16bf;
typedef __attribute__((ext_vector_type(8)))  float  v8f;

#define DDIM    256
#define KCODES  4096
#define NROWS   65536
#define HW      1024
#define MBLOCK  256                 // rows of x per block
#define NCHUNK  (KCODES / 16)       // 16-code chunks
#define LPAD    (DDIM + 8)          // LDS row stride in bf16 (528B, 16B-aligned)

union Frag16 { v16bf v; uint4 q[2]; };

// ---------------- kernel 1: embedding f32 -> bf16 + |e|^2 ------------------
__global__ __launch_bounds__(256)
void prep_embed_kernel(const float* __restrict__ emb,
                       bf16* __restrict__ eb,
                       float* __restrict__ enorm)
{
    const int k = blockIdx.x;
    const int t = threadIdx.x;
    float v = emb[k * DDIM + t];
    eb[k * DDIM + t] = (bf16)v;
    float s = v * v;
    #pragma unroll
    for (int off = 16; off >= 1; off >>= 1) s += __shfl_xor(s, off, 32);
    __shared__ float ps[8];
    if ((t & 31) == 0) ps[t >> 5] = s;
    __syncthreads();
    if (t == 0) {
        float tot = 0.f;
        #pragma unroll
        for (int i = 0; i < 8; ++i) tot += ps[i];
        enorm[k] = tot;
    }
}

// ---------------- kernel 2: zero histogram ---------------------------------
__global__ __launch_bounds__(256)
void zero_counts_kernel(unsigned* __restrict__ counts)
{
    counts[blockIdx.x * 256 + threadIdx.x] = 0u;
}

// ---------------- kernel 3: WMMA distance + argmin -------------------------
__global__ __launch_bounds__(256)
void vq_argmin_kernel(const float* __restrict__ x,
                      const bf16*  __restrict__ eb,
                      const float* __restrict__ enorm,
                      int* __restrict__ idxOut)
{
    __shared__ __align__(16) bf16  lA[MBLOCK][LPAD];   // x tile, [row][d]
    __shared__ __align__(16) bf16  lB[2][16][LPAD];    // codebook chunk, dbl-buf
    __shared__ float lEn[2][16];

    const int tid  = threadIdx.x;
    const int lane = tid & 31;
    const int wave = tid >> 5;
    const int lcol = lane & 15;
    const int half = lane >> 4;

    // ---- stage & transpose x tile: memory is [img][d][hw], want [row][d] ----
    const int img = blockIdx.x >> 2;          // 1024 rows per image
    const int hw0 = (blockIdx.x & 3) << 8;    // 256 rows per block
    {
        const unsigned base = ((unsigned)img * DDIM) * HW + (unsigned)hw0 + tid;
        #pragma unroll 4
        for (int d = 0; d < DDIM; d += 2) {
            float v0 = x[base + (unsigned)d * HW];        // coalesced over tid
            float v1 = x[base + (unsigned)(d + 1) * HW];
            union { bf16 h[2]; unsigned u; } p;
            p.h[0] = (bf16)v0; p.h[1] = (bf16)v1;
            *(unsigned*)&lA[tid][d] = p.u;
        }
    }
    __syncthreads();

    // ---- A fragments register-resident: 2 row-sets x 8 k-steps per wave ----
    // 16-bit A 16x32 layout: lanes 0-15 row M=lane, K = 32s+{0..7,16..23};
    //                        lanes 16-31 same rows, K = 32s+{8..15,24..31}.
    Frag16 fa[2][8];
    #pragma unroll
    for (int t = 0; t < 2; ++t) {
        const int row = (wave << 5) + (t << 4) + lcol;
        #pragma unroll
        for (int s = 0; s < 8; ++s) {
            fa[t][s].q[0] = *(const uint4*)&lA[row][32 * s + 8 * half];
            fa[t][s].q[1] = *(const uint4*)&lA[row][32 * s + 16 + 8 * half];
        }
    }

    float bb0[8], bb1[8];
    int   bi0[8], bi1[8];
    #pragma unroll
    for (int j = 0; j < 8; ++j) {
        bb0[j] = 3.4e38f; bb1[j] = 3.4e38f; bi0[j] = 0; bi1[j] = 0;
    }

    // ---- prime chunk 0 of the codebook into LDS buffer 0 ----
    const int rB  = tid >> 4;   // code row within chunk
    const int seg = tid & 15;   // 16-element (32B) segment of D
    {
        const uint4* g = (const uint4*)&eb[rB * DDIM + seg * 16];
        uint4 q0 = g[0], q1 = g[1];
        uint4* dst = (uint4*)&lB[0][rB][seg * 16];
        dst[0] = q0; dst[1] = q1;
        if (tid < 16) lEn[0][tid] = enorm[tid];
    }
    __syncthreads();

    // ---- main loop over 256 chunks of 16 codes ----
    #pragma unroll 1
    for (int c = 0; c < NCHUNK; ++c) {
        const int buf  = c & 1;
        const bool more = (c + 1 < NCHUNK);

        // issue next chunk's global loads now; store to LDS after compute
        uint4 nq0, nq1; float nen = 0.f;
        if (more) {
            const uint4* g =
                (const uint4*)&eb[((c + 1) * 16 + rB) * DDIM + seg * 16];
            nq0 = g[0]; nq1 = g[1];
            if (tid < 16) nen = enorm[(c + 1) * 16 + tid];
        }

        // Preload ALL 8 B fragments of this chunk into a dedicated register
        // array so every ds_load_b128 issues before the WMMA chain starts:
        // the first WMMA then waits only on the oldest outstanding pair and
        // the rest of the LDS latency drains behind the matrix pipe.
        // B 32x16 layout: lanes 0-15 col=lane K=32s+{0..7,16..23};
        //                 lanes 16-31 col=lane-16 K=32s+{8..15,24..31}.
        Frag16 fb[8];
        #pragma unroll
        for (int s = 0; s < 8; ++s) {
            fb[s].q[0] = *(const uint4*)&lB[buf][lcol][32 * s + 8 * half];
            fb[s].q[1] = *(const uint4*)&lB[buf][lcol][32 * s + 16 + 8 * half];
        }

        // 16 chained WMMAs: dot(x_row, e_code) over D=256
        v8f acc0 = {0.f,0.f,0.f,0.f,0.f,0.f,0.f,0.f};
        v8f acc1 = {0.f,0.f,0.f,0.f,0.f,0.f,0.f,0.f};
        #pragma unroll
        for (int s = 0; s < 8; ++s) {
            acc0 = __builtin_amdgcn_wmma_f32_16x16x32_bf16(
                       false, fa[0][s].v, false, fb[s].v, (short)0, acc0, false, false);
            acc1 = __builtin_amdgcn_wmma_f32_16x16x32_bf16(
                       false, fa[1][s].v, false, fb[s].v, (short)0, acc1, false, false);
        }

        // dist proxy = |e|^2 - 2 x.e ; running argmin in C/D layout
        const float en   = lEn[buf][lcol];
        const int   code = (c << 4) + lcol;
        #pragma unroll
        for (int j = 0; j < 8; ++j) {
            float c0 = en - 2.0f * acc0[j];
            float c1 = en - 2.0f * acc1[j];
            if (c0 < bb0[j]) { bb0[j] = c0; bi0[j] = code; }
            if (c1 < bb1[j]) { bb1[j] = c1; bi1[j] = code; }
        }

        if (more) {
            uint4* dst = (uint4*)&lB[buf ^ 1][rB][seg * 16];
            dst[0] = nq0; dst[1] = nq1;
            if (tid < 16) lEn[buf ^ 1][tid] = nen;
        }
        __syncthreads();
    }

    // ---- cross-lane min+index reduction (each 16-lane half = one row set) --
    const int rowBase = (blockIdx.x << 8) + (wave << 5);
    #pragma unroll
    for (int j = 0; j < 8; ++j) {
        float v = bb0[j]; int ix = bi0[j];
        #pragma unroll
        for (int off = 8; off >= 1; off >>= 1) {
            float v2 = __shfl_xor(v, off, 32);
            int   i2 = __shfl_xor(ix, off, 32);
            if (v2 < v || (v2 == v && i2 < ix)) { v = v2; ix = i2; }
        }
        if (lcol == 0) idxOut[rowBase + (half << 3) + j] = ix;

        v = bb1[j]; ix = bi1[j];
        #pragma unroll
        for (int off = 8; off >= 1; off >>= 1) {
            float v2 = __shfl_xor(v, off, 32);
            int   i2 = __shfl_xor(ix, off, 32);
            if (v2 < v || (v2 == v && i2 < ix)) { v = v2; ix = i2; }
        }
        if (lcol == 0) idxOut[rowBase + 16 + (half << 3) + j] = ix;
    }
}

// ---------------- kernel 4: gather + output + loss partials + histogram ----
__global__ __launch_bounds__(256)
void gather_out_kernel(const float* __restrict__ x,
                       const float* __restrict__ emb,
                       const int*   __restrict__ idx,
                       float* __restrict__ out,
                       float* __restrict__ partial,
                       unsigned* __restrict__ counts)
{
    const int tid = threadIdx.x;
    const int i   = blockIdx.x * 256 + tid;       // index into [B,C,H,W]
    const int b   = i >> 18;                      // C*HW = 2^18
    const int d   = (i >> 10) & 255;
    const int hw  = i & 1023;
    const int n   = (b << 10) | hw;

    const int   k  = idx[n];
    const float e  = emb[k * DDIM + d];
    const float xv = x[i];
    out[i] = e;                                   // straight-through == gather
    if (d == 0) atomicAdd(&counts[k], 1u);        // one vote per row n

    const float df = e - xv;
    __shared__ float red[256];
    red[tid] = df * df;
    __syncthreads();
    #pragma unroll
    for (int s = 128; s > 0; s >>= 1) {
        if (tid < s) red[tid] += red[tid + s];
        __syncthreads();
    }
    if (tid == 0) partial[blockIdx.x] = red[0];
}

// ---------------- kernel 5: fixed-order final reduction --------------------
__global__ __launch_bounds__(256)
void finalize_kernel(const float* __restrict__ partial,
                     const unsigned* __restrict__ counts,
                     float* __restrict__ scal)   // scal[0]=loss, scal[1]=ppl
{
    const int t = threadIdx.x;
    __shared__ float red[256];

    float s = 0.f;
    for (int j = 0; j < 256; ++j) s += partial[t * 256 + j];
    red[t] = s;
    __syncthreads();
    #pragma unroll
    for (int st = 128; st > 0; st >>= 1) {
        if (t < st) red[t] += red[t + st];
        __syncthreads();
    }
    const float loss = 1.25f * red[0] * (1.0f / 16777216.0f);
    __syncthreads();

    float h = 0.f;
    for (int j = 0; j < 16; ++j) {
        float p = (float)counts[t * 16 + j] * (1.0f / 65536.0f);
        h += p * logf(p + 1e-10f);
    }
    red[t] = h;
    __syncthreads();
    #pragma unroll
    for (int st = 128; st > 0; st >>= 1) {
        if (t < st) red[t] += red[t + st];
        __syncthreads();
    }
    if (t == 0) { scal[0] = loss; scal[1] = expf(-red[0]); }
}

// ---------------------------------------------------------------------------
extern "C" void kernel_launch(void* const* d_in, const int* in_sizes, int n_in,
                              void* d_out, int out_size, void* d_ws, size_t ws_size,
                              hipStream_t stream)
{
    const float* x   = (const float*)d_in[0];   // 64*256*32*32
    const float* emb = (const float*)d_in[1];   // 4096*256
    float* out = (float*)d_out;

    char* ws = (char*)d_ws;
    bf16*     eb      = (bf16*)ws;                                   // 2 MB
    float*    enorm   = (float*)(ws + 2097152);                      // 16 KB
    int*      idx     = (int*)(ws + 2097152 + 16384);                // 256 KB
    unsigned* counts  = (unsigned*)(ws + 2097152 + 16384 + 262144);  // 16 KB
    float*    partial = (float*)(ws + 2097152 + 16384 + 262144 + 16384); // 256 KB

    prep_embed_kernel <<<KCODES, 256, 0, stream>>>(emb, eb, enorm);
    zero_counts_kernel<<<KCODES / 256, 256, 0, stream>>>(counts);
    vq_argmin_kernel  <<<NROWS / MBLOCK, 256, 0, stream>>>(x, eb, enorm, idx);
    gather_out_kernel <<<(NROWS * DDIM) / 256, 256, 0, stream>>>(x, emb, idx, out,
                                                                 partial, counts);
    finalize_kernel   <<<1, 256, 0, stream>>>(partial, counts,
                                              out + (out_size - 2));
}
